// ZippyPoint_90632399880639
// MI455X (gfx1250) — 8-bit matrix operand order repaired and checked by reading, compiled, not run on hardware
//
#include <hip/hip_runtime.h>
#include <hip/hip_bf16.h>

typedef __attribute__((ext_vector_type(8))) int v8i;

__global__ void k_quant(const float* __restrict__ x, signed char* __restrict__ q,
                        size_t n, float inv_s, int binarize) {
  size_t i = (size_t)blockIdx.x * 256 + threadIdx.x;
  if (i >= n) return;
  float v = x[i];
  if (binarize) {
    q[i] = (v >= 0.f) ? (signed char)1 : (signed char)-1;
  } else {
    float r = __builtin_rintf(v * inv_s);
    r = fminf(fmaxf(r, -127.f), 127.f);
    q[i] = (signed char)(int)r;
  }
}

__global__ void k_wstats(const float* __restrict__ w, int khwcin, int cout, int mode,
                         const float* g, const float* be, const float* mu, const float* va,
                         const float* bias, float s_act,
                         float* __restrict__ s_w, float* __restrict__ Ac,
                         float* __restrict__ Bc) {
  int co = blockIdx.x;
  __shared__ float red[256];
  float acc = 0.f;
  for (int i = threadIdx.x; i < khwcin; i += 256) {
    float a = fabsf(w[(size_t)i * cout + co]);
    acc = (mode == 1) ? acc + a : fmaxf(acc, a);
  }
  red[threadIdx.x] = acc;
  __syncthreads();
  for (int s = 128; s > 0; s >>= 1) {
    if (threadIdx.x < s) {
      float o = red[threadIdx.x + s];
      red[threadIdx.x] = (mode == 1) ? red[threadIdx.x] + o : fmaxf(red[threadIdx.x], o);
    }
    __syncthreads();
  }
  if (threadIdx.x == 0) {
    float sw = (mode == 1) ? red[0] / (float)khwcin
             : (mode == 0) ? red[0] / 127.f + 1e-8f
                           : 1.f;
    s_w[co] = sw;
    float a = s_act * sw, b = 0.f;
    if (g) { float bn = g[co] * rsqrtf(va[co] + 1e-3f); b = be[co] - mu[co] * bn; a *= bn; }
    if (bias) b += bias[co];
    Ac[co] = a; Bc[co] = b;
  }
}

__global__ void k_wpack(const float* __restrict__ w, const float* __restrict__ s_w,
                        signed char* __restrict__ pk,
                        int KHW, int Cin, int Cout, int nchunk, int mode) {
  size_t total = (size_t)(Cout >> 4) * KHW * nchunk * 1024;
  size_t i = (size_t)blockIdx.x * 256 + threadIdx.x;
  if (i >= total) return;
  int byte = i & 3, reg = (i >> 2) & 7, lane = (i >> 5) & 31;
  size_t r = i >> 10;
  int j = (int)(r % nchunk); r /= nchunk;
  int tap = (int)(r % KHW);
  int grp = (int)(r / KHW);
  int n = lane & 15;
  int k = ((reg >= 4) ? 32 : 0) + ((lane >= 16) ? 16 : 0) + ((reg & 3) << 2) + byte;
  int ci = (j << 6) + k;
  signed char qv = 0;
  if (ci < Cin) {
    int co = (grp << 4) + n;
    float wv = w[((size_t)tap * Cin + ci) * Cout + co];
    if (mode == 1) qv = (wv >= 0.f) ? (signed char)1 : (signed char)-1;
    else {
      float q = __builtin_rintf(wv / s_w[co]);
      q = fminf(fmaxf(q, -127.f), 127.f);
      qv = (signed char)(int)q;
    }
  }
  pk[i] = qv;
}

__global__ __launch_bounds__(128)
void k_conv_wmma(const signed char* __restrict__ act, const signed char* __restrict__ wpk,
                 const float* __restrict__ Ac, const float* __restrict__ Bc,
                 const float* __restrict__ resid, float* __restrict__ outp,
                 int B, int H, int W, int Cin, int Cout, int KH, int KW,
                 int relu1, int hsw2) {
  __shared__ int Atile[2][16 * 16];
  const int KHW = KH * KW;
  const int nchunk = (Cin + 63) >> 6;
  const int niter = KHW * nchunk;
  const int nblk = (Cout + 255) >> 8;
  const bool use_async = (KHW == 1) && ((Cin & 63) == 0);

  const int tile_x = blockIdx.x << 4;
  const int oy = blockIdx.y;
  const int b = blockIdx.z / nblk;
  const int zgrp = blockIdx.z % nblk;
  const int lane = threadIdx.x & 31;
  const int wvid = __builtin_amdgcn_readfirstlane(threadIdx.x >> 5);
  const int gbase = zgrp * 16 + wvid * 4;
  const int half = lane >> 4, row = lane & 15;

  const int sm = threadIdx.x >> 3;
  const int sb = (threadIdx.x & 7) << 3;

  v8i acc[4];
  #pragma unroll
  for (int q = 0; q < 4; ++q) acc[q] = (v8i){0, 0, 0, 0, 0, 0, 0, 0};

  auto stage = [&](int it) {
    const int tap = it / nchunk, j = it - tap * nchunk;
    const int iy = oy + tap / KW - (KH >> 1);
    const int ix = tile_x + sm + tap % KW - (KW >> 1);
    const int c = (j << 6) + sb;
    const int sel = it & 1;
    if (use_async) {
      unsigned lofs = (unsigned)(size_t)((char*)&Atile[sel][0] + (sm << 6) + sb);
      unsigned long long ga =
          (unsigned long long)(act + ((((size_t)b * H + iy) * W + ix) * Cin + c));
      asm volatile("global_load_async_to_lds_b64 %0, %1, off"
                   :: "v"(lofs), "v"(ga) : "memory");
    } else {
      unsigned long long v = 0ull;
      if (iy >= 0 && iy < H && ix >= 0 && ix < W && c < Cin)
        v = *(const unsigned long long*)(act + ((((size_t)b * H + iy) * W + ix) * Cin + c));
      *(unsigned long long*)((char*)&Atile[sel][0] + (sm << 6) + sb) = v;
    }
  };

  stage(0);
  for (int it = 0; it < niter; ++it) {
    if (use_async) asm volatile("s_wait_asynccnt 0" ::: "memory");
    __syncthreads();
    if (it + 1 < niter) stage(it + 1);
    const int sel = it & 1;
    v8i Ar;
    #pragma unroll
    for (int r = 0; r < 8; ++r) {
      int kb = ((r >> 2) << 5) + (half << 4) + ((r & 3) << 2);
      Ar[r] = Atile[sel][(row << 4) + (kb >> 2)];
    }
    const int tap = it / nchunk, j = it - tap * nchunk;
    #pragma unroll
    for (int q = 0; q < 4; ++q) {
      if ((gbase + q) * 16 < Cout) {
        const int* bp = (const int*)(wpk +
            ((((size_t)(gbase + q) * KHW + tap) * nchunk + j) << 10) + (lane << 5));
        __builtin_prefetch(bp + 256, 0, 1);
        v8i Br;
        #pragma unroll
        for (int r = 0; r < 8; ++r) Br[r] = bp[r];
        acc[q] = __builtin_amdgcn_wmma_i32_16x16x64_iu8(true, Ar, true, Br, acc[q],
                                                        false, false);
      }
    }
  }

  __builtin_amdgcn_s_wait_tensorcnt(0);

  const int n = lane & 15;
  #pragma unroll
  for (int q = 0; q < 4; ++q) {
    const int gq = gbase + q;
    if (gq * 16 < Cout) {
      const int co = (gq << 4) + n;
      const float a = Ac[co], bb = Bc[co];
      #pragma unroll
      for (int r = 0; r < 8; ++r) {
        const int m = r + (half << 3);
        float v = (float)acc[q][r] * a + bb;
        if (relu1) v = fmaxf(v, 0.f);
        size_t oi = (((size_t)b * H + oy) * W + tile_x + m) * Cout + co;
        if (resid) v += resid[oi];
        if (hsw2) { float t = fminf(fmaxf(v + 3.f, 0.f), 6.f); v = v * t * (1.f / 6.f); }
        outp[oi] = v;
      }
    }
  }
}

__global__ void k_conv_scalar(const float* __restrict__ in, const float* __restrict__ w,
                              const float* __restrict__ s_w, const float* __restrict__ Ac,
                              const float* __restrict__ Bc, float* __restrict__ outp,
                              int B, int H, int W, int Cin, int Cout, int KH, int KW,
                              int qflag, float inv_sa, int actmode, int border) {
  size_t total = (size_t)B * H * W * Cout;
  size_t i = (size_t)blockIdx.x * 256 + threadIdx.x;
  if (i >= total) return;
  int co = (int)(i % Cout); size_t r = i / Cout;
  int x = (int)(r % W); r /= W;
  int y = (int)(r % H); int b = (int)(r / H);
  float invsw = qflag ? 1.f / s_w[co] : 1.f;
  float acc = 0.f;
  for (int ky = 0; ky < KH; ++ky) {
    int iy = y + ky - KH / 2; if (iy < 0 || iy >= H) continue;
    for (int kx = 0; kx < KW; ++kx) {
      int ix = x + kx - KW / 2; if (ix < 0 || ix >= W) continue;
      const float* ip = in + (((size_t)b * H + iy) * W + ix) * Cin;
      const float* wp = w + (size_t)(ky * KW + kx) * Cin * Cout + co;
      for (int ci = 0; ci < Cin; ++ci) {
        float a = ip[ci], wvv = wp[(size_t)ci * Cout];
        if (qflag) {
          a = fminf(fmaxf(__builtin_rintf(a * inv_sa), -127.f), 127.f);
          wvv = fminf(fmaxf(__builtin_rintf(wvv * invsw), -127.f), 127.f);
        }
        acc += a * wvv;
      }
    }
  }
  float v = acc * Ac[co] + Bc[co];
  if (actmode == 1) { float t = fminf(fmaxf(v + 3.f, 0.f), 6.f); v = v * t * (1.f / 6.f); }
  else if (actmode == 2) { v = 1.f / (1.f + __expf(-v)); }
  else if (actmode == 3) { v = tanhf(v); }
  if (border && (x == 0 || y == 0 || x == W - 1 || y == H - 1)) v = 0.f;
  outp[i] = v;
}

__global__ void k_s2d(const float* __restrict__ in, float* __restrict__ outp,
                      int B, int H, int W, int C) {
  size_t total = (size_t)B * H * W * C;
  size_t i = (size_t)blockIdx.x * 256 + threadIdx.x;
  if (i >= total) return;
  int C4 = 4 * C, W2 = W / 2, H2 = H / 2;
  int c4 = (int)(i % C4); size_t r = i / C4;
  int w2 = (int)(r % W2); r /= W2;
  int h2 = (int)(r % H2); int b = (int)(r / H2);
  int ii = c4 / (2 * C); int rem = c4 % (2 * C); int jj = rem / C; int c = rem % C;
  outp[i] = in[(((size_t)b * H + 2 * h2 + ii) * W + 2 * w2 + jj) * C + c];
}

__global__ void k_d2s(const float* __restrict__ in, float* __restrict__ outp,
                      int B, int H, int W, int C) {
  int Co = C / 4, H2 = H * 2, W2 = W * 2;
  size_t total = (size_t)B * H2 * W2 * Co;
  size_t i = (size_t)blockIdx.x * 256 + threadIdx.x;
  if (i >= total) return;
  int co = (int)(i % Co); size_t r = i / Co;
  int x = (int)(r % W2); r /= W2;
  int y = (int)(r % H2); int b = (int)(r / H2);
  int h = y >> 1, ii = y & 1, w = x >> 1, jj = x & 1;
  outp[i] = in[(((size_t)b * H + h) * W + w) * C + (ii * 2 + jj) * Co + co];
}

__global__ void k_map2grid(const float* __restrict__ p, float* __restrict__ pos,
                           int B, int Hc, int Wc) {
  size_t total = (size_t)B * Hc * Wc;
  size_t i = (size_t)blockIdx.x * 256 + threadIdx.x;
  if (i >= total) return;
  int x = (int)(i % Wc);
  int y = (int)((i / Wc) % Hc);
  pos[i * 2]     = ((float)x + 0.5f) * 8.f + p[i * 2]     * 8.f;
  pos[i * 2 + 1] = ((float)y + 0.5f) * 8.f + p[i * 2 + 1] * 8.f;
}

__global__ void k_sample_sign(const float* __restrict__ d, const float* __restrict__ pos,
                              float* __restrict__ outp,
                              int B, int Hc, int Wc, int Hd, int Wd, int C) {
  int pix = blockIdx.x;
  int c = threadIdx.x;
  int b = pix / (Hc * Wc);
  float cx = pos[(size_t)pix * 2], cy = pos[(size_t)pix * 2 + 1];
  float px = cx * (float)(Wd - 1) / (float)(Wc * 8 - 1);
  float py = cy * (float)(Hd - 1) / (float)(Hc * 8 - 1);
  px = fminf(fmaxf(px, 0.f), (float)(Wd - 1));
  py = fminf(fmaxf(py, 0.f), (float)(Hd - 1));
  float x0f = floorf(px), y0f = floorf(py);
  int x0 = (int)x0f, y0 = (int)y0f;
  int x1 = min(x0 + 1, Wd - 1), y1 = min(y0 + 1, Hd - 1);
  float wx = px - x0f, wy = py - y0f;
  const float* base = d + (size_t)b * Hd * Wd * C;
  float v00 = base[((size_t)y0 * Wd + x0) * C + c];
  float v01 = base[((size_t)y0 * Wd + x1) * C + c];
  float v10 = base[((size_t)y1 * Wd + x0) * C + c];
  float v11 = base[((size_t)y1 * Wd + x1) * C + c];
  float v = (v00 * (1.f - wx) + v01 * wx) * (1.f - wy) + (v10 * (1.f - wx) + v11 * wx) * wy;
  outp[(size_t)pix * C + c] = (v >= 0.f) ? 1.f : -1.f;
}

struct Layer { const float *w, *g, *be, *mu, *va, *b, *wsc; };

static inline void* wsalloc(char* base, size_t& off, size_t bytes) {
  void* p = base + off;
  off = (off + bytes + 255) & ~(size_t)255;
  return p;
}

static void run_wmma_conv(hipStream_t st, char* base, size_t& off, signed char* qbuf,
                          const float* in, float* outp, const float* resid,
                          const Layer& L, bool use_wsc,
                          int Bn, int H, int W, int Cin, int Cout, int KH, int KW,
                          int mode, float qmax, int relu1, int hsw2) {
  int khw = KH * KW, nchunk = (Cin + 63) / 64, G = Cout / 16;
  float* s_w = (float*)wsalloc(base, off, (size_t)Cout * 4);
  float* Ac  = (float*)wsalloc(base, off, (size_t)Cout * 4);
  float* Bc  = (float*)wsalloc(base, off, (size_t)Cout * 4);
  size_t pkbytes = (size_t)G * khw * nchunk * 1024;
  signed char* pk = (signed char*)wsalloc(base, off, pkbytes);
  const float* w  = use_wsc ? L.wsc : L.w;
  const float* g  = use_wsc ? nullptr : L.g;
  const float* be = use_wsc ? nullptr : L.be;
  const float* mu = use_wsc ? nullptr : L.mu;
  const float* va = use_wsc ? nullptr : L.va;
  const float* bs = use_wsc ? nullptr : L.b;
  float s_act = (mode == 1) ? 1.f : qmax / 127.f;
  k_wstats<<<Cout, 256, 0, st>>>(w, khw * Cin, Cout, mode, g, be, mu, va, bs, s_act,
                                 s_w, Ac, Bc);
  k_wpack<<<(unsigned)((pkbytes + 255) / 256), 256, 0, st>>>(w, s_w, pk, khw, Cin, Cout,
                                                             nchunk, mode);
  size_t nel = (size_t)Bn * H * W * Cin;
  k_quant<<<(unsigned)((nel + 255) / 256), 256, 0, st>>>(in, qbuf, nel,
                                                         (mode == 1) ? 1.f : 127.f / qmax,
                                                         mode == 1);
  dim3 grid(W / 16, H, Bn * ((Cout + 255) / 256));
  k_conv_wmma<<<grid, 128, 0, st>>>(qbuf, pk, Ac, Bc, resid, outp,
                                    Bn, H, W, Cin, Cout, KH, KW, relu1, hsw2);
}

static void run_scalar_conv(hipStream_t st, char* base, size_t& off,
                            const float* in, float* outp, const Layer& L,
                            int Bn, int H, int W, int Cin, int Cout, int KH, int KW,
                            int qflag, float qmax, int actmode, int border) {
  float* s_w = (float*)wsalloc(base, off, (size_t)Cout * 4);
  float* Ac  = (float*)wsalloc(base, off, (size_t)Cout * 4);
  float* Bc  = (float*)wsalloc(base, off, (size_t)Cout * 4);
  float s_act = qflag ? qmax / 127.f : 1.f;
  k_wstats<<<Cout, 256, 0, st>>>(L.w, KH * KW * Cin, Cout, qflag ? 0 : 2,
                                 L.g, L.be, L.mu, L.va, L.b, s_act, s_w, Ac, Bc);
  size_t total = (size_t)Bn * H * W * Cout;
  k_conv_scalar<<<(unsigned)((total + 255) / 256), 256, 0, st>>>(
      in, L.w, s_w, Ac, Bc, outp, Bn, H, W, Cin, Cout, KH, KW,
      qflag, qflag ? 127.f / qmax : 1.f, actmode, border);
}

extern "C" void kernel_launch(void* const* d_in, const int* in_sizes, int n_in,
                              void* d_out, int out_size, void* d_ws, size_t ws_size,
                              hipStream_t stream) {
  (void)in_sizes; (void)n_in; (void)out_size; (void)ws_size;
  const int Bn = 4;
  const float* x = (const float*)d_in[0];

  int pi = 1;
  auto getf = [&]() -> const float* { return (const float*)d_in[pi++]; };
  auto next = [&](bool bn, bool bias, bool sc) -> Layer {
    Layer L{}; L.w = getf();
    if (bn) { L.g = getf(); L.be = getf(); L.mu = getf(); L.va = getf(); }
    if (bias) L.b = getf();
    if (sc) L.wsc = getf();
    return L;
  };
  Layer Lin = next(true, false, false);
  Layer Lp1 = next(true, false, false);
  Layer L1a = next(true, false, false);
  Layer L1b = next(true, false, false);
  Layer Lp2 = next(true, false, false);
  Layer L2a = next(true, false, false);
  Layer L2b = next(true, false, false);
  Layer Lp3 = next(true, false, false);
  Layer L3a = next(true, false, false);
  Layer L3b = next(true, false, false);
  Layer L4a = next(true, false, true);
  Layer L4b = next(true, false, false);
  Layer LSa = next(true, false, false);
  Layer LSb = next(false, true, false);
  Layer LPa = next(true, false, false);
  Layer LPb = next(false, true, false);
  Layer LDa = next(true, false, false);
  Layer LDb = next(true, false, false);
  Layer LDc = next(true, false, false);
  Layer LDd = next(false, true, false);

  char* base = (char*)d_ws;
  size_t off = 0;
  const size_t NMAX = 39321600ull;
  float* bufA = (float*)wsalloc(base, off, NMAX * 4);
  float* bufB = (float*)wsalloc(base, off, NMAX * 4);
  float* bufC = (float*)wsalloc(base, off, NMAX * 4);
  signed char* qbuf = (signed char*)wsalloc(base, off, NMAX);

  float* out_sc  = (float*)d_out;
  float* out_pos = (float*)d_out + 19200;
  float* out_d   = (float*)d_out + 57600;

  auto ew = [&](size_t n) { return (unsigned)((n + 255) / 256); };

  run_scalar_conv(stream, base, off, x, bufA, Lin, Bn, 480, 640, 3, 32, 3, 3, 1, 1.0f, 1, 0);
  k_s2d<<<ew((size_t)Bn * 480 * 640 * 32), 256, 0, stream>>>(bufA, bufB, Bn, 480, 640, 32);
  run_wmma_conv(stream, base, off, qbuf, bufB, bufA, nullptr, Lp1, false,
                Bn, 240, 320, 128, 32, 1, 1, 0, 1.0f, 0, 1);
  run_wmma_conv(stream, base, off, qbuf, bufA, bufB, bufA, L1a, false,
                Bn, 240, 320, 32, 32, 3, 3, 1, 1.0f, 1, 0);
  run_wmma_conv(stream, base, off, qbuf, bufB, bufA, bufB, L1b, false,
                Bn, 240, 320, 32, 32, 3, 3, 1, 1.0f, 1, 1);
  k_s2d<<<ew((size_t)Bn * 240 * 320 * 32), 256, 0, stream>>>(bufA, bufB, Bn, 240, 320, 32);
  run_wmma_conv(stream, base, off, qbuf, bufB, bufA, nullptr, Lp2, false,
                Bn, 120, 160, 128, 64, 1, 1, 0, 6.0f, 0, 1);
  run_wmma_conv(stream, base, off, qbuf, bufA, bufB, bufA, L2a, false,
                Bn, 120, 160, 64, 64, 3, 3, 1, 1.0f, 1, 0);
  run_wmma_conv(stream, base, off, qbuf, bufB, bufA, bufB, L2b, false,
                Bn, 120, 160, 64, 64, 3, 3, 1, 1.0f, 1, 1);
  k_s2d<<<ew((size_t)Bn * 120 * 160 * 64), 256, 0, stream>>>(bufA, bufB, Bn, 120, 160, 64);
  run_wmma_conv(stream, base, off, qbuf, bufB, bufA, nullptr, Lp3, false,
                Bn, 60, 80, 256, 128, 1, 1, 0, 6.0f, 0, 1);
  run_wmma_conv(stream, base, off, qbuf, bufA, bufB, bufA, L3a, false,
                Bn, 60, 80, 128, 128, 3, 3, 1, 1.0f, 1, 0);
  run_wmma_conv(stream, base, off, qbuf, bufB, bufA, bufB, L3b, false,
                Bn, 60, 80, 128, 128, 3, 3, 1, 1.0f, 1, 1);
  run_wmma_conv(stream, base, off, qbuf, bufA, bufC, nullptr, L4a, true,
                Bn, 60, 80, 128, 256, 1, 1, 0, 6.0f, 0, 0);
  run_wmma_conv(stream, base, off, qbuf, bufA, bufB, bufC, L4a, false,
                Bn, 60, 80, 128, 256, 3, 3, 1, 1.0f, 1, 0);
  run_wmma_conv(stream, base, off, qbuf, bufB, bufA, bufB, L4b, false,
                Bn, 60, 80, 256, 256, 3, 3, 1, 1.0f, 1, 1);
  run_wmma_conv(stream, base, off, qbuf, bufA, bufB, nullptr, LSa, false,
                Bn, 60, 80, 256, 256, 3, 3, 0, 6.0f, 0, 1);
  run_scalar_conv(stream, base, off, bufB, out_sc, LSb, Bn, 60, 80, 256, 1, 3, 3, 0, 1.f, 2, 1);
  run_wmma_conv(stream, base, off, qbuf, bufA, bufB, nullptr, LPa, false,
                Bn, 60, 80, 256, 256, 3, 3, 0, 6.0f, 0, 1);
  run_scalar_conv(stream, base, off, bufB, bufC, LPb, Bn, 60, 80, 256, 2, 3, 3, 0, 1.f, 3, 0);
  k_map2grid<<<ew((size_t)Bn * 60 * 80), 256, 0, stream>>>(bufC, out_pos, Bn, 60, 80);
  run_wmma_conv(stream, base, off, qbuf, bufA, bufB, nullptr, LDa, false,
                Bn, 60, 80, 256, 256, 3, 3, 0, 6.0f, 0, 1);
  run_wmma_conv(stream, base, off, qbuf, bufB, bufC, nullptr, LDb, false,
                Bn, 60, 80, 256, 512, 3, 3, 0, 6.0f, 0, 0);
  k_d2s<<<ew((size_t)Bn * 120 * 160 * 128), 256, 0, stream>>>(bufC, bufB, Bn, 60, 80, 512);
  run_wmma_conv(stream, base, off, qbuf, bufB, bufC, nullptr, LDc, false,
                Bn, 120, 160, 128, 256, 3, 3, 0, 6.0f, 0, 1);
  run_wmma_conv(stream, base, off, qbuf, bufC, bufB, nullptr, LDd, false,
                Bn, 120, 160, 256, 256, 3, 3, 0, 6.0f, 0, 0);
  k_sample_sign<<<Bn * 60 * 80, 256, 0, stream>>>(bufB, out_pos, out_d,
                                                  Bn, 60, 80, 120, 160, 256);
}
